// NeuralTransformer_39608188404105
// MI455X (gfx1250) — matrix operand order repaired and checked by reading, not compiled, not run on hardware
//
#include <hip/hip_runtime.h>
#include <hip/hip_bf16.h>
#include <cstdint>

// ---------------------------------------------------------------------------
// MI455X (gfx1250) implementation. Compute-bound (~186 GFLOP vs ~0.4GB moved):
// all six GEMMs run on v_wmma_f32_16x16x32_f16 (wave32).
// Token GEMMs (178/186 GFLOP): 128x128 C-tile, 8 waves, 32x64 per wave,
// double-buffered LDS (1 barrier per K-step, B-fragment reused across 2 WMMAs).
// Attention GEMMs: guarded 64x64 tiles (marginal FLOPs, ragged 208/197 edges).
// ---------------------------------------------------------------------------

typedef _Float16 h16_t;
typedef __attribute__((ext_vector_type(16))) _Float16 v16h;
typedef __attribute__((ext_vector_type(8)))  float    v8f;
typedef __attribute__((ext_vector_type(4)))  float    vf4;   // POD 16B vector

#define BATCH 64
#define SEQ   197
#define NP    208          // padded sequence (13*16)
#define DIM   768
#define HEADS 12
#define HDIM  64
#define MLPD  3072
#define TOKENS (BATCH*SEQ) // 12608
#define SCALE_Q 0.125f     // 64^-0.5

union FragU { vf4 f4[2]; v16h v; };

__device__ __forceinline__ float wave_red_sum(float v) {
#pragma unroll
  for (int o = 16; o > 0; o >>= 1) v += __shfl_xor(v, o, 32);
  return v;
}
__device__ __forceinline__ float wave_red_max(float v) {
#pragma unroll
  for (int o = 16; o > 0; o >>= 1) v = fmaxf(v, __shfl_xor(v, o, 32));
  return v;
}

// ---------------------------------------------------------------------------
// Token GEMM:  C[M][Nc] = A[M][K] @ W[Nc][K]^T
// 256 threads / 8 waves; C tile 128x128; wave (wr,wc) owns rows wr*32..+31,
// cols wc*64..+63 (2x4 accumulators). Double-buffered LDS, one barrier/K-step.
// ---------------------------------------------------------------------------
template <class Epi>
__global__ __launch_bounds__(256) void gemm_tok(const h16_t* __restrict__ A,
                                                const h16_t* __restrict__ W,
                                                int M, int K, Epi epi) {
  constexpr int LDT = 40;                     // halves; pad to spread LDS banks
  __shared__ h16_t As[2][128 * LDT];
  __shared__ h16_t Bs[2][128 * LDT];
  const int mBase = blockIdx.x * 128;
  const int nBase = blockIdx.y * 128;
  const int t = threadIdx.x;
  const int lane = t & 31, wave = t >> 5;
  const int wr = wave & 3;                    // row strip (32 rows)
  const int wc = wave >> 2;                   // col strip (64 cols)
  const int frow = t >> 1, fseg = (t & 1) * 16;  // global->LDS fill coords
  const int kbA = (lane >> 4) * 8;            // A frag K-base (interleaved)
  const int kbB = (lane >> 4) * 8;            // B frag K-base (interleaved)
  const int mrow = mBase + frow;
  const bool aok = mrow < M;

  v8f acc[2][4] = {};
  vf4 ra0, ra1, rb0, rb1;

  auto gload = [&](int k0) {
    vf4 z = {0.f, 0.f, 0.f, 0.f};
    if (aok) {
      const vf4* ga = (const vf4*)(A + (size_t)mrow * K + k0 + fseg);
      ra0 = ga[0]; ra1 = ga[1];
    } else { ra0 = z; ra1 = z; }
    const vf4* gb = (const vf4*)(W + (size_t)(nBase + frow) * K + k0 + fseg);
    rb0 = gb[0]; rb1 = gb[1];
  };
  auto lstore = [&](int buf) {
    *(vf4*)(&As[buf][frow * LDT + fseg])     = ra0;
    *(vf4*)(&As[buf][frow * LDT + fseg + 8]) = ra1;
    *(vf4*)(&Bs[buf][frow * LDT + fseg])     = rb0;
    *(vf4*)(&Bs[buf][frow * LDT + fseg + 8]) = rb1;
  };
  auto compute = [&](int buf) {
    FragU af[2];
#pragma unroll
    for (int i = 0; i < 2; ++i) {
      const int row = wr * 32 + i * 16 + (lane & 15);
      af[i].f4[0] = *(const vf4*)(&As[buf][row * LDT + kbA]);
      af[i].f4[1] = *(const vf4*)(&As[buf][row * LDT + kbA + 16]);
    }
#pragma unroll
    for (int j = 0; j < 4; ++j) {
      FragU bf;
      const int nl = wc * 64 + j * 16 + (lane & 15);
      bf.f4[0] = *(const vf4*)(&Bs[buf][nl * LDT + kbB]);
      bf.f4[1] = *(const vf4*)(&Bs[buf][nl * LDT + kbB + 16]);
      acc[0][j] = __builtin_amdgcn_wmma_f32_16x16x32_f16(false, af[0].v, false, bf.v,
                                                         (short)0, acc[0][j], false, false);
      acc[1][j] = __builtin_amdgcn_wmma_f32_16x16x32_f16(false, af[1].v, false, bf.v,
                                                         (short)0, acc[1][j], false, false);
    }
  };

  gload(0);
  lstore(0);
  __syncthreads();
  int cur = 0;
  for (int k0 = 32; k0 < K; k0 += 32) {
    gload(k0);          // prefetch next tile into registers
    compute(cur);       // 8 WMMAs from current LDS buffers
    lstore(cur ^ 1);    // publish next tile
    __syncthreads();
    cur ^= 1;
  }
  compute(cur);

  const int mr  = mBase + wr * 32 + ((lane >> 4) << 3);
  const int nc0 = nBase + wc * 64 + (lane & 15);
#pragma unroll
  for (int i = 0; i < 2; ++i)
#pragma unroll
    for (int j = 0; j < 4; ++j)
#pragma unroll
      for (int r = 0; r < 8; ++r) {
        const int m = mr + i * 16 + r;
        if (m < M) epi(m, nc0 + j * 16, acc[i][j][r]);
      }
}

// ---------------------------------------------------------------------------
// Per-(b,h) attention GEMM with edge guards (M/N limit 208, partial K).
// 128 threads / 4 waves; C tile 64x64; wave owns a 16x64 strip.
// ---------------------------------------------------------------------------
template <class Epi>
__global__ __launch_bounds__(128) void gemm_attn(const h16_t* __restrict__ Ab, unsigned long long sA, int lda,
                                                 const h16_t* __restrict__ Bb, unsigned long long sB, int ldb,
                                                 int kTot, int kVal, int mLim, int nLim, Epi epi) {
  constexpr int LDT = 40;
  __shared__ h16_t As[64 * LDT];
  __shared__ h16_t Bs[64 * LDT];
  const int bh = blockIdx.z;
  const h16_t* A = Ab + (size_t)bh * sA;
  const h16_t* B = Bb + (size_t)bh * sB;
  const int mBase = blockIdx.x * 64;
  const int nBase = blockIdx.y * 64;
  const int t = threadIdx.x;
  const int lane = t & 31, wave = t >> 5;
  const int frow = t >> 1, fseg = (t & 1) * 16;
  const int mloc = wave * 16 + (lane & 15);
  const int kbA = (lane >> 4) * 8;
  const int kbB = (lane >> 4) * 8;

  v8f acc[4] = {};
  for (int k0 = 0; k0 < kTot; k0 += 32) {
    const int kc = k0 + fseg;
    const bool kok = kc < kVal;
    vf4 z = {0.f, 0.f, 0.f, 0.f};
    vf4 a0 = z, a1 = z, b0 = z, b1 = z;
    if (kok && (mBase + frow) < mLim) {
      const vf4* ga = (const vf4*)(A + (size_t)(mBase + frow) * lda + kc);
      a0 = ga[0]; a1 = ga[1];
    }
    if (kok && (nBase + frow) < nLim) {
      const vf4* gb = (const vf4*)(B + (size_t)(nBase + frow) * ldb + kc);
      b0 = gb[0]; b1 = gb[1];
    }
    *(vf4*)(&As[frow * LDT + fseg])     = a0;
    *(vf4*)(&As[frow * LDT + fseg + 8]) = a1;
    *(vf4*)(&Bs[frow * LDT + fseg])     = b0;
    *(vf4*)(&Bs[frow * LDT + fseg + 8]) = b1;
    __syncthreads();
    FragU af;
    af.f4[0] = *(const vf4*)(&As[mloc * LDT + kbA]);
    af.f4[1] = *(const vf4*)(&As[mloc * LDT + kbA + 16]);
#pragma unroll
    for (int j = 0; j < 4; ++j) {
      FragU bf;
      const int nl = j * 16 + (lane & 15);
      bf.f4[0] = *(const vf4*)(&Bs[nl * LDT + kbB]);
      bf.f4[1] = *(const vf4*)(&Bs[nl * LDT + kbB + 16]);
      acc[j] = __builtin_amdgcn_wmma_f32_16x16x32_f16(false, af.v, false, bf.v,
                                                      (short)0, acc[j], false, false);
    }
    __syncthreads();
  }
  const int mr  = mBase + wave * 16 + ((lane >> 4) << 3);
  const int nc0 = nBase + (lane & 15);
#pragma unroll
  for (int j = 0; j < 4; ++j)
#pragma unroll
    for (int r = 0; r < 8; ++r) epi(bh, mr + r, nc0 + j * 16, acc[j][r]);
}

// ------------------------------ epilogues ---------------------------------
struct EpiQKV {
  const float *qb, *vb, *sc, *sh;
  h16_t *q, *k, *vT;
  __device__ void operator()(int m, int n, float a) const {
    float bias = (n < DIM) ? qb[n] : ((n >= 2 * DIM) ? vb[n - 2 * DIM] : 0.f);
    float v = (a + bias) * sc[n] + sh[n];
    int b = m / SEQ, tok = m - b * SEQ;
    int which = n / DIM, d = n - which * DIM, hh = d >> 6, hd = d & 63;
    size_t bh = (size_t)b * HEADS + hh;
    if (which == 0)      q[(bh * NP + tok) * HDIM + hd] = (h16_t)(v * SCALE_Q);
    else if (which == 1) k[(bh * NP + tok) * HDIM + hd] = (h16_t)v;
    else                 vT[(bh * HDIM + hd) * NP + tok] = (h16_t)v;  // V stored transposed
  }
};
struct EpiScores {
  h16_t* s;
  __device__ void operator()(int bh, int m, int n, float a) const {
    if (m < NP && n < NP) s[((size_t)bh * NP + m) * NP + n] = (h16_t)a;
  }
};
struct EpiAV {
  h16_t* y;
  __device__ void operator()(int bh, int m, int n, float a) const {
    if (m < SEQ) {
      int b = bh / HEADS, hh = bh - b * HEADS;
      y[((size_t)(b * SEQ + m)) * DIM + hh * HDIM + n] = (h16_t)a;
    }
  }
};
struct EpiProj {
  const float *bias, *sc, *sh, *gamma, *x;
  float* x1;
  __device__ void operator()(int m, int n, float a) const {
    float v = (a + bias[n]) * sc[n] + sh[n];
    size_t i = (size_t)m * DIM + n;
    x1[i] = x[i] + gamma[n] * v;
  }
};
struct EpiFC1 {
  const float* bias; h16_t* out;
  __device__ void operator()(int m, int n, float a) const {
    float tt = a + bias[n];
    out[(size_t)m * MLPD + n] = (h16_t)(0.5f * tt * (1.f + erff(tt * 0.70710678118f)));
  }
};
struct EpiFC2 {
  const float *bias, *sc, *sh, *gamma, *x1;
  float* out;
  __device__ void operator()(int m, int n, float a) const {
    float v = (a + bias[n]) * sc[n] + sh[n];
    size_t i = (size_t)m * DIM + n;
    out[i] = x1[i] + gamma[n] * v;
  }
};

// ------------------------- elementwise kernels ----------------------------
__global__ void cvtf16(const float* __restrict__ s, h16_t* __restrict__ d, int n) {
  int i = blockIdx.x * 256 + threadIdx.x;
  if (i < n) d[i] = (h16_t)s[i];
}

__global__ __launch_bounds__(256) void ln_ssf(const float* __restrict__ x,
                                              const float* __restrict__ w, const float* __restrict__ b,
                                              const float* __restrict__ sc, const float* __restrict__ sh,
                                              h16_t* __restrict__ out) {
  const int row = blockIdx.x;
  const float* xr = x + (size_t)row * DIM;
  const int t = threadIdx.x;
  __shared__ float red[8];
  float v0 = xr[t], v1 = xr[t + 256], v2 = xr[t + 512];
  float s = wave_red_sum(v0 + v1 + v2);
  if ((t & 31) == 0) red[t >> 5] = s;
  __syncthreads();
  float mean = 0.f;
#pragma unroll
  for (int i = 0; i < 8; ++i) mean += red[i];
  mean *= (1.f / (float)DIM);
  __syncthreads();
  float d0 = v0 - mean, d1 = v1 - mean, d2 = v2 - mean;
  float q = wave_red_sum(d0 * d0 + d1 * d1 + d2 * d2);
  if ((t & 31) == 0) red[t >> 5] = q;
  __syncthreads();
  float var = 0.f;
#pragma unroll
  for (int i = 0; i < 8; ++i) var += red[i];
  var *= (1.f / (float)DIM);
  float r = rsqrtf(var + 1e-5f);
  h16_t* o = out + (size_t)row * DIM;
  o[t]       = (h16_t)((d0 * r * w[t]       + b[t])       * sc[t]       + sh[t]);
  o[t + 256] = (h16_t)((d1 * r * w[t + 256] + b[t + 256]) * sc[t + 256] + sh[t + 256]);
  o[t + 512] = (h16_t)((d2 * r * w[t + 512] + b[t + 512]) * sc[t + 512] + sh[t + 512]);
}

__global__ __launch_bounds__(128) void softmax_bias(h16_t* __restrict__ attn,
                                                    const float* __restrict__ table,
                                                    const int* __restrict__ relidx) {
  const int n = blockIdx.x, bh = blockIdx.y, h = bh % HEADS;
  h16_t* row = attn + ((size_t)bh * NP + n) * NP;
  const int t = threadIdx.x;
  if (n >= SEQ) {  // zero padded rows (keeps AV A-operand clean)
    for (int c = t; c < NP; c += 128) row[c] = (h16_t)0.f;
    return;
  }
  __shared__ float red[4];
  const int c0 = t, c1 = t + 128;
  float e0 = -1e30f, e1 = -1e30f;
  if (c0 < SEQ) e0 = (float)row[c0] + table[relidx[n * SEQ + c0] * HEADS + h];
  if (c1 < SEQ) e1 = (float)row[c1] + table[relidx[n * SEQ + c1] * HEADS + h];
  float mx = wave_red_max(fmaxf(e0, e1));
  if ((t & 31) == 0) red[t >> 5] = mx;
  __syncthreads();
  mx = fmaxf(fmaxf(red[0], red[1]), fmaxf(red[2], red[3]));
  __syncthreads();
  e0 = (c0 < SEQ) ? __expf(e0 - mx) : 0.f;
  e1 = (c1 < SEQ) ? __expf(e1 - mx) : 0.f;
  float sm = wave_red_sum(e0 + e1);
  if ((t & 31) == 0) red[t >> 5] = sm;
  __syncthreads();
  const float inv = 1.f / (red[0] + red[1] + red[2] + red[3]);
  row[c0] = (h16_t)(e0 * inv);
  if (c1 < NP) row[c1] = (h16_t)(e1 * inv);
}

__global__ __launch_bounds__(256) void headmix(const h16_t* __restrict__ attn,
                                               h16_t* __restrict__ mixed,
                                               const float* __restrict__ bc) {
  const int b = blockIdx.y;
  const size_t pos = (size_t)blockIdx.x * 256 + threadIdx.x;  // < 208*208 = 43264 (169*256)
  const size_t plane = (size_t)NP * NP;
  const size_t base = (size_t)b * HEADS * plane + pos;
  float av[HEADS];
#pragma unroll
  for (int hh = 0; hh < HEADS; ++hh) av[hh] = (float)attn[base + hh * plane];
#pragma unroll
  for (int k = 0; k < HEADS; ++k) {
    float s = av[k];  // identity term of (bases_coeff^T + I)
#pragma unroll
    for (int hh = 0; hh < HEADS; ++hh) s += av[hh] * bc[k * HEADS + hh];
    mixed[base + k * plane] = (h16_t)s;
  }
}

// ------------------------------- launch -----------------------------------
extern "C" void kernel_launch(void* const* d_in, const int* in_sizes, int n_in,
                              void* d_out, int out_size, void* d_ws, size_t ws_size,
                              hipStream_t stream) {
  const float* x       = (const float*)d_in[0];
  const float* w_qkv   = (const float*)d_in[1];
  const float* q_bias  = (const float*)d_in[2];
  const float* v_bias  = (const float*)d_in[3];
  const float* ssc_qkv = (const float*)d_in[4];
  const float* ssh_qkv = (const float*)d_in[5];
  const float* table   = (const float*)d_in[6];
  const float* bcoeff  = (const float*)d_in[7];
  const float* w_proj  = (const float*)d_in[8];
  const float* b_proj  = (const float*)d_in[9];
  const float* ssc_pr  = (const float*)d_in[10];
  const float* ssh_pr  = (const float*)d_in[11];
  const float* n1w     = (const float*)d_in[12];
  const float* n1b     = (const float*)d_in[13];
  const float* n2w     = (const float*)d_in[14];
  const float* n2b     = (const float*)d_in[15];
  const float* s1c     = (const float*)d_in[16];
  const float* s1h     = (const float*)d_in[17];
  const float* s2c     = (const float*)d_in[18];
  const float* s2h     = (const float*)d_in[19];
  const float* w_fc1   = (const float*)d_in[20];
  const float* b_fc1   = (const float*)d_in[21];
  const float* w_fc2   = (const float*)d_in[22];
  const float* b_fc2   = (const float*)d_in[23];
  const float* smc     = (const float*)d_in[24];
  const float* smh     = (const float*)d_in[25];
  const float* g1      = (const float*)d_in[26];
  const float* g2      = (const float*)d_in[27];
  const int*   relidx  = (const int*)d_in[28];
  float* out = (float*)d_out;
  (void)in_sizes; (void)n_in; (void)out_size; (void)ws_size;

  char* ws = (char*)d_ws;
  size_t off = 0;
  auto alloc = [&](size_t bytes) -> char* {
    char* p = ws + off; off += (bytes + 255) & ~(size_t)255; return p;
  };
  h16_t* wq16   = (h16_t*)alloc((size_t)3 * DIM * DIM * 2);
  h16_t* wp16   = (h16_t*)alloc((size_t)DIM * DIM * 2);
  h16_t* wf1_16 = (h16_t*)alloc((size_t)MLPD * DIM * 2);
  h16_t* wf2_16 = (h16_t*)alloc((size_t)DIM * MLPD * 2);
  h16_t* h16v   = (h16_t*)alloc((size_t)TOKENS * DIM * 2);
  const size_t qkvBytes = (size_t)BATCH * HEADS * NP * HDIM * 2;  // 20.45 MB each
  h16_t* q16  = (h16_t*)alloc(qkvBytes);
  h16_t* k16  = (h16_t*)alloc(qkvBytes);
  h16_t* vT16 = (h16_t*)alloc(qkvBytes);
  float* x1   = (float*)alloc((size_t)TOKENS * DIM * 4);
  const size_t attnBytes = (size_t)BATCH * HEADS * NP * NP * 2;   // 66.45 MB each
  h16_t* scr = (h16_t*)alloc(attnBytes);
  h16_t* mix = (h16_t*)alloc(attnBytes);
  h16_t* m16 = scr;   // reuse scores+mixed region for MLP hidden (77.5MB <= 133MB)
  h16_t* y16 = q16;   // reuse q buffer for attention output (19.4MB <= 20.4MB)

  const int mg = (TOKENS + 127) / 128;  // 99 M-tiles (last partially masked)

  // 1) weights fp32 -> f16 (once per launch; deterministic)
  cvtf16<<<(3 * DIM * DIM + 255) / 256, 256, 0, stream>>>(w_qkv, wq16, 3 * DIM * DIM);
  cvtf16<<<(DIM * DIM + 255) / 256, 256, 0, stream>>>(w_proj, wp16, DIM * DIM);
  cvtf16<<<(MLPD * DIM + 255) / 256, 256, 0, stream>>>(w_fc1, wf1_16, MLPD * DIM);
  cvtf16<<<(DIM * MLPD + 255) / 256, 256, 0, stream>>>(w_fc2, wf2_16, DIM * MLPD);
  // 2) LN1 + SSF -> h (f16)
  ln_ssf<<<TOKENS, 256, 0, stream>>>(x, n1w, n1b, s1c, s1h, h16v);
  // 3) zero q/k/vT so padded rows/cols are clean (contiguous in ws)
  hipMemsetAsync(q16, 0, qkvBytes * 3, stream);
  // 4) QKV GEMM (+bias, SSF, q-scale, scatter to q/k/vT)
  gemm_tok<<<dim3(mg, (3 * DIM) / 128), 256, 0, stream>>>(
      h16v, wq16, TOKENS, DIM, EpiQKV{q_bias, v_bias, ssc_qkv, ssh_qkv, q16, k16, vT16});
  // 5) scores = q @ k^T per (b,h)
  gemm_attn<<<dim3(4, 4, BATCH * HEADS), 128, 0, stream>>>(
      q16, (unsigned long long)NP * HDIM, HDIM,
      k16, (unsigned long long)NP * HDIM, HDIM,
      64, 64, NP, NP, EpiScores{scr});
  // 6) softmax(scores + rel_bias)  (in place)
  softmax_bias<<<dim3(NP, BATCH * HEADS), 128, 0, stream>>>(scr, table, relidx);
  // 7) DCF head mixing: mixed[b,k] = attn[b,k] + sum_h attn[b,h]*bases_coeff[k,h]
  headmix<<<dim3((NP * NP) / 256, BATCH), 256, 0, stream>>>(scr, mix, bcoeff);
  // 8) y = mixed @ v  per (b,h)   (K padded 208->224 with zero fill)
  gemm_attn<<<dim3(4, 1, BATCH * HEADS), 128, 0, stream>>>(
      mix, (unsigned long long)NP * NP, NP,
      vT16, (unsigned long long)HDIM * NP, NP,
      224, NP, NP, HDIM, EpiAV{y16});
  // 9) proj + SSF + gamma1 residual -> x1 (f32)
  gemm_tok<<<dim3(mg, DIM / 128), 256, 0, stream>>>(
      y16, wp16, TOKENS, DIM, EpiProj{b_proj, ssc_pr, ssh_pr, g1, x, x1});
  // 10) LN2 + SSF -> h (f16, reuse)
  ln_ssf<<<TOKENS, 256, 0, stream>>>(x1, n2w, n2b, s2c, s2h, h16v);
  // 11) FC1 + exact GELU -> m16 (f16)
  gemm_tok<<<dim3(mg, MLPD / 128), 256, 0, stream>>>(
      h16v, wf1_16, TOKENS, DIM, EpiFC1{b_fc1, m16});
  // 12) FC2 + SSF + gamma2 residual -> out (f32)
  gemm_tok<<<dim3(mg, DIM / 128), 256, 0, stream>>>(
      m16, wf2_16, TOKENS, MLPD, EpiFC2{b_fc2, smc, smh, g2, x1, out});
}
